// multihead_cross_attention_50216757624952
// MI455X (gfx1250) — matrix operand order repaired and checked by reading, not compiled, not run on hardware
//
#include <hip/hip_runtime.h>
#include <math.h>
#include <stdint.h>

#define BB  4
#define SS  2048
#define DD  1024
#define HH  16
#define DKK 64
#define HDK (HH*DKK)   // 1024

typedef __attribute__((ext_vector_type(16))) _Float16 v16h;
typedef __attribute__((ext_vector_type(8)))  _Float16 v8h;
typedef __attribute__((ext_vector_type(4)))  _Float16 v4h;
typedef __attribute__((ext_vector_type(8)))  float    v8f;
typedef __attribute__((ext_vector_type(4)))  unsigned int v4u;
typedef __attribute__((ext_vector_type(8)))  int      v8i;
typedef __attribute__((ext_vector_type(4)))  int      v4i;

__device__ __forceinline__ v16h cat16(v8h lo, v8h hi) {
  return __builtin_shufflevector(lo, hi, 0,1,2,3,4,5,6,7,8,9,10,11,12,13,14,15);
}

__device__ __forceinline__ v8f wmma_f16(v16h a, v16h b, v8f c) {
  return __builtin_amdgcn_wmma_f32_16x16x32_f16(false, a, false, b, (short)0, c, false, false);
}

// A fragment (16 x 32, MxK) from LDS, rows contiguous in K, row stride in halfs.
__device__ __forceinline__ v16h ld_afrag(const _Float16* base, int stride, int lane) {
  const int row = lane & 15, kb = (lane >> 4) * 8;
  const _Float16* p = base + row * stride + kb;
  v8h lo = *(const v8h*)(p);
  v8h hi = *(const v8h*)(p + 16);
  return cat16(lo, hi);
}

// B fragment (32 x 16, KxN) from LDS stored transposed: row = n, contiguous in K.
__device__ __forceinline__ v16h ld_bfrag(const _Float16* base, int stride, int lane) {
  const int n = lane & 15, kb = (lane >> 4) * 8;
  const _Float16* p = base + n * stride + kb;
  v8h lo = *(const v8h*)(p);
  v8h hi = *(const v8h*)(p + 16);
  return cat16(lo, hi);
}

// ---- Tensor Data Mover: 2D tile (rows x 32 dwords) global -> LDS, with HW row
// padding of 8 dwords every 32 dwords (matches the 80-half LDS row stride).
// D# packed per CDNA5 ISA; clang-23 6-arg builtin.
__device__ __forceinline__ void tdm_load_2d(uint32_t lds_off, const void* gaddr,
                                            int rows, int stride_dwords) {
  const uint64_t ga = (uint64_t)(uintptr_t)gaddr;
  v4u g0;
  g0[0] = 1u;                                             // count=1, user mode
  g0[1] = lds_off;                                        // LDS byte address
  g0[2] = (uint32_t)ga;                                   // global_addr[31:0]
  g0[3] = (uint32_t)((ga >> 32) & 0x01FFFFFFu) | (2u << 30); // addr[56:32] | type=2
  v8i g1;
  // data_size=4B, pad_enable, pad_interval=32 dwords (code 4), pad_amount=8 dwords (code 7)
  g1[0] = (int)((2u << 16) | (1u << 20) | (4u << 22) | (7u << 25));
  g1[1] = (int)(32u << 16);                               // tensor_dim0 = 32 dwords
  g1[2] = (int)(((uint32_t)rows & 0xFFFFu) << 16);        // tensor_dim1 = rows
  g1[3] = (int)(32u << 16);                               // tile_dim0 = 32 dwords
  g1[4] = (int)((uint32_t)rows & 0xFFFFu);                // tile_dim1 = rows, tile_dim2=0
  g1[5] = stride_dwords;                                  // tensor_dim0_stride[31:0]
  g1[6] = 0;
  g1[7] = 0;
  const v4i z4 = {0, 0, 0, 0};
  const v8i z8 = {0, 0, 0, 0, 0, 0, 0, 0};
  __builtin_amdgcn_tensor_load_to_lds(g0, g1, z4, z4, z8, 0);
}

// -------- Projection GEMM: dst = f16(X[M,1024] @ W[N,1024]^T + bias) ----------
// vmode 0: dst layout [B,H,S,DK]  (Q, K);  vmode 1: [B,H,DK,S]  (V transposed)
__global__ __launch_bounds__(256) void proj_kernel(const float* __restrict__ X,
                                                   const float* __restrict__ W,
                                                   const float* __restrict__ bias,
                                                   _Float16* __restrict__ dst,
                                                   int vmode) {
  __shared__ __align__(16) _Float16 As[128 * 48];
  __shared__ __align__(16) _Float16 Bs[64 * 48];
  const int tid = threadIdx.x;
  const int lane = tid & 31, wv = tid >> 5;
  const int ll = lane & 15, lg = lane >> 4;
  const int m0 = blockIdx.x * 128;
  const int n0 = blockIdx.y * 64;

  v8f acc[4] = {};

  for (int k0 = 0; k0 < DD; k0 += 32) {
    float4 fa[4], fb[2];
#pragma unroll
    for (int i = 0; i < 4; ++i) {
      const int idx = tid + i * 256;
      const int row = idx >> 3, c4 = (idx & 7) << 2;
      fa[i] = *(const float4*)(X + (size_t)(m0 + row) * DD + k0 + c4);
    }
#pragma unroll
    for (int i = 0; i < 2; ++i) {
      const int idx = tid + i * 256;
      const int row = idx >> 3, c4 = (idx & 7) << 2;
      fb[i] = *(const float4*)(W + (size_t)(n0 + row) * DD + k0 + c4);
    }
    if (k0 + 32 < DD) {  // prefetch next K-chunk
      __builtin_prefetch(X + (size_t)(m0 + (tid >> 3)) * DD + k0 + 32, 0, 3);
      __builtin_prefetch(W + (size_t)(n0 + (tid >> 3 & 63)) * DD + k0 + 32, 0, 3);
    }
#pragma unroll
    for (int i = 0; i < 4; ++i) {
      const int idx = tid + i * 256;
      const int row = idx >> 3, c4 = (idx & 7) << 2;
      v4h h; h[0] = (_Float16)fa[i].x; h[1] = (_Float16)fa[i].y;
             h[2] = (_Float16)fa[i].z; h[3] = (_Float16)fa[i].w;
      *(v4h*)&As[row * 48 + c4] = h;
    }
#pragma unroll
    for (int i = 0; i < 2; ++i) {
      const int idx = tid + i * 256;
      const int row = idx >> 3, c4 = (idx & 7) << 2;
      v4h h; h[0] = (_Float16)fb[i].x; h[1] = (_Float16)fb[i].y;
             h[2] = (_Float16)fb[i].z; h[3] = (_Float16)fb[i].w;
      *(v4h*)&Bs[row * 48 + c4] = h;
    }
    __syncthreads();
    const v16h a  = ld_afrag(&As[wv * 16 * 48], 48, lane);
    const v16h b0 = ld_bfrag(&Bs[0 * 16 * 48], 48, lane);
    const v16h b1 = ld_bfrag(&Bs[1 * 16 * 48], 48, lane);
    const v16h b2 = ld_bfrag(&Bs[2 * 16 * 48], 48, lane);
    const v16h b3 = ld_bfrag(&Bs[3 * 16 * 48], 48, lane);
    acc[0] = wmma_f16(a, b0, acc[0]);
    acc[1] = wmma_f16(a, b1, acc[1]);
    acc[2] = wmma_f16(a, b2, acc[2]);
    acc[3] = wmma_f16(a, b3, acc[3]);
    __syncthreads();
  }

#pragma unroll
  for (int t = 0; t < 4; ++t) {
    const int n = n0 + t * 16 + ll;
    const float bv = bias[n];
    const int h = n >> 6, dk = n & 63;
#pragma unroll
    for (int r = 0; r < 8; ++r) {
      const int m = m0 + wv * 16 + r + (lg << 3);
      const int b = m >> 11, s = m & 2047;       // m = b*S + s
      const float v = acc[t][r] + bv;
      size_t off;
      if (vmode) off = ((size_t)(b * HH + h) * DKK + dk) * SS + s;
      else       off = ((size_t)(b * HH + h) * SS + s) * DKK + dk;
      dst[off] = (_Float16)v;
    }
  }
}

// -------- Flash attention: per (b,h), 64 query rows per block, online softmax --
__global__ __launch_bounds__(128) void flash_attn_kernel(const _Float16* __restrict__ Q,
                                                         const _Float16* __restrict__ K,
                                                         const _Float16* __restrict__ Vt,
                                                         _Float16* __restrict__ ctx) {
  __shared__ __align__(16) _Float16 Qs[64 * 80];
  __shared__ __align__(16) _Float16 Ks[64 * 80];
  __shared__ __align__(16) _Float16 Vs[64 * 80];
  __shared__ __align__(16) _Float16 Ps[64 * 80];
  const int tid = threadIdx.x;
  const int lane = tid & 31, wv = tid >> 5;
  const int ll = lane & 15, lg = lane >> 4;
  const int q0 = blockIdx.x * 64;
  const int bh = blockIdx.y;                     // b*H + h

  const _Float16* Qg = Q  + ((size_t)bh * SS + q0) * DKK;  // 64x64 contiguous
  const _Float16* Kg = K  + (size_t)bh * SS * DKK;
  const _Float16* Vg = Vt + (size_t)bh * DKK * SS;

  // Low 32 bits of a flat LDS pointer are the LDS byte offset (aperture layout).
  const uint32_t qs_off = (uint32_t)(uintptr_t)(void*)Qs;
  const uint32_t ks_off = (uint32_t)(uintptr_t)(void*)Ks;
  const uint32_t vs_off = (uint32_t)(uintptr_t)(void*)Vs;

  if (wv == 0) {
    tdm_load_2d(qs_off, Qg, 64, 32);             // Q tile: 64 rows, contiguous
    __builtin_amdgcn_s_wait_tensorcnt(0);
  }
  __syncthreads();

  const v16h qa0 = ld_afrag(&Qs[wv * 16 * 80], 80, lane);       // dk 0..31
  const v16h qa1 = ld_afrag(&Qs[wv * 16 * 80 + 32], 80, lane);  // dk 32..63

  v8f oacc[4] = {};
  float mst[8], lst[8];
#pragma unroll
  for (int r = 0; r < 8; ++r) { mst[r] = -3.0e38f; lst[r] = 0.f; }

  const float scale = 0.125f;  // 1/sqrt(64)

  for (int j = 0; j < SS / 64; ++j) {
    if (wv == 0) {
      tdm_load_2d(ks_off, Kg + (size_t)j * 64 * DKK, 64, 32);    // 64 keys x 64 dk
      tdm_load_2d(vs_off, Vg + (size_t)j * 64, 64, SS / 2);      // 64 dk x 64 keys, stride S halfs
      __builtin_amdgcn_s_wait_tensorcnt(0);
    }
    __syncthreads();

    // scores tile: S = Q (16x64) @ K^T  -> 4 key sub-tiles
    v8f sac[4] = {};
#pragma unroll
    for (int t = 0; t < 4; ++t) {
      const v16h b0 = ld_bfrag(&Ks[t * 16 * 80], 80, lane);
      const v16h b1 = ld_bfrag(&Ks[t * 16 * 80 + 32], 80, lane);
      sac[t] = wmma_f16(qa0, b0, sac[t]);
      sac[t] = wmma_f16(qa1, b1, sac[t]);
    }

    // online softmax; each lane owns 8 rows (r + 8*lg), cols ll in 4 sub-tiles
    float corr[8];
#pragma unroll
    for (int r = 0; r < 8; ++r) {
      float mloc = -3.0e38f;
#pragma unroll
      for (int t = 0; t < 4; ++t) { sac[t][r] *= scale; mloc = fmaxf(mloc, sac[t][r]); }
#pragma unroll
      for (int off = 1; off < 16; off <<= 1) mloc = fmaxf(mloc, __shfl_xor(mloc, off));
      const float mnew = fmaxf(mst[r], mloc);
      const float c = __expf(mst[r] - mnew);
      float rsum = 0.f;
#pragma unroll
      for (int t = 0; t < 4; ++t) {
        const float p = __expf(sac[t][r] - mnew);
        sac[t][r] = p;
        rsum += p;
      }
#pragma unroll
      for (int off = 1; off < 16; off <<= 1) rsum += __shfl_xor(rsum, off);
      lst[r] = lst[r] * c + rsum;
      mst[r] = mnew;
      corr[r] = c;
    }
#pragma unroll
    for (int u = 0; u < 4; ++u)
#pragma unroll
      for (int r = 0; r < 8; ++r) oacc[u][r] *= corr[r];

    // re-shape P from C-layout to A-frag layout through LDS (wave-private rows)
    _Float16* Pw = &Ps[wv * 16 * 80];
#pragma unroll
    for (int t = 0; t < 4; ++t)
#pragma unroll
      for (int r = 0; r < 8; ++r)
        Pw[(r + (lg << 3)) * 80 + t * 16 + ll] = (_Float16)sac[t][r];
    __syncthreads();

    const v16h pa0 = ld_afrag(Pw, 80, lane);        // keys 0..31
    const v16h pa1 = ld_afrag(Pw + 32, 80, lane);   // keys 32..63
#pragma unroll
    for (int u = 0; u < 4; ++u) {
      const v16h b0 = ld_bfrag(&Vs[u * 16 * 80], 80, lane);
      const v16h b1 = ld_bfrag(&Vs[u * 16 * 80 + 32], 80, lane);
      oacc[u] = wmma_f16(pa0, b0, oacc[u]);
      oacc[u] = wmma_f16(pa1, b1, oacc[u]);
    }
    __syncthreads();
  }

  // normalize and store context as [B, S, H*DK] f16
  const int b = bh >> 4, h = bh & 15;
#pragma unroll
  for (int u = 0; u < 4; ++u) {
#pragma unroll
    for (int r = 0; r < 8; ++r) {
      const int s = q0 + wv * 16 + r + (lg << 3);
      const int dk = u * 16 + ll;
      const float v = oacc[u][r] / lst[r];
      ctx[(size_t)(b * SS + s) * HDK + h * DKK + dk] = (_Float16)v;
    }
  }
}

// -------- Output GEMM: out[M,1024] = ctx_f16[M,1024] @ Wo[1024,1024]^T (f32 out)
__global__ __launch_bounds__(256) void outproj_kernel(const _Float16* __restrict__ Xh,
                                                      const float* __restrict__ W,
                                                      float* __restrict__ out) {
  __shared__ __align__(16) _Float16 As[128 * 48];
  __shared__ __align__(16) _Float16 Bs[64 * 48];
  const int tid = threadIdx.x;
  const int lane = tid & 31, wv = tid >> 5;
  const int ll = lane & 15, lg = lane >> 4;
  const int m0 = blockIdx.x * 128;
  const int n0 = blockIdx.y * 64;

  v8f acc[4] = {};

  for (int k0 = 0; k0 < HDK; k0 += 32) {
    v8h ha[2];
    float4 fb[2];
#pragma unroll
    for (int i = 0; i < 2; ++i) {   // 128x32 halfs
      const int idx = tid + i * 256;
      const int row = idx >> 2, c8 = (idx & 3) << 3;
      ha[i] = *(const v8h*)(Xh + (size_t)(m0 + row) * HDK + k0 + c8);
    }
#pragma unroll
    for (int i = 0; i < 2; ++i) {   // 64x32 f32
      const int idx = tid + i * 256;
      const int row = idx >> 3, c4 = (idx & 7) << 2;
      fb[i] = *(const float4*)(W + (size_t)(n0 + row) * HDK + k0 + c4);
    }
    if (k0 + 32 < HDK) {
      __builtin_prefetch(Xh + (size_t)(m0 + (tid >> 2)) * HDK + k0 + 32, 0, 3);
      __builtin_prefetch(W + (size_t)(n0 + (tid >> 3 & 63)) * HDK + k0 + 32, 0, 3);
    }
#pragma unroll
    for (int i = 0; i < 2; ++i) {
      const int idx = tid + i * 256;
      const int row = idx >> 2, c8 = (idx & 3) << 3;
      *(v8h*)&As[row * 48 + c8] = ha[i];
    }
#pragma unroll
    for (int i = 0; i < 2; ++i) {
      const int idx = tid + i * 256;
      const int row = idx >> 3, c4 = (idx & 7) << 2;
      v4h h; h[0] = (_Float16)fb[i].x; h[1] = (_Float16)fb[i].y;
             h[2] = (_Float16)fb[i].z; h[3] = (_Float16)fb[i].w;
      *(v4h*)&Bs[row * 48 + c4] = h;
    }
    __syncthreads();
    const v16h a  = ld_afrag(&As[wv * 16 * 48], 48, lane);
    const v16h b0 = ld_bfrag(&Bs[0 * 16 * 48], 48, lane);
    const v16h b1 = ld_bfrag(&Bs[1 * 16 * 48], 48, lane);
    const v16h b2 = ld_bfrag(&Bs[2 * 16 * 48], 48, lane);
    const v16h b3 = ld_bfrag(&Bs[3 * 16 * 48], 48, lane);
    acc[0] = wmma_f16(a, b0, acc[0]);
    acc[1] = wmma_f16(a, b1, acc[1]);
    acc[2] = wmma_f16(a, b2, acc[2]);
    acc[3] = wmma_f16(a, b3, acc[3]);
    __syncthreads();
  }

#pragma unroll
  for (int t = 0; t < 4; ++t) {
    const int n = n0 + t * 16 + ll;
#pragma unroll
    for (int r = 0; r < 8; ++r) {
      const int m = m0 + wv * 16 + r + (lg << 3);
      out[(size_t)m * DD + n] = acc[t][r];
    }
  }
}

extern "C" void kernel_launch(void* const* d_in, const int* in_sizes, int n_in,
                              void* d_out, int out_size, void* d_ws, size_t ws_size,
                              hipStream_t stream) {
  (void)in_sizes; (void)n_in; (void)out_size; (void)ws_size;
  const float* Xq = (const float*)d_in[0];
  const float* Xk = (const float*)d_in[1];
  const float* Xv = (const float*)d_in[2];
  const float* Wq = (const float*)d_in[3];
  const float* bq = (const float*)d_in[4];
  const float* Wk = (const float*)d_in[5];
  const float* bk = (const float*)d_in[6];
  const float* Wv = (const float*)d_in[7];
  const float* bv = (const float*)d_in[8];
  const float* Wo = (const float*)d_in[9];
  float* out = (float*)d_out;

  _Float16* ws = (_Float16*)d_ws;
  const size_t QN = (size_t)BB * HH * SS * DKK;   // 8M halfs = 16 MB each
  _Float16* Qh  = ws;
  _Float16* Kh  = ws + QN;
  _Float16* Vth = ws + 2 * QN;
  _Float16* Ctx = ws + 3 * QN;

  const dim3 gblk(64, 16), blk(256);
  proj_kernel<<<gblk, blk, 0, stream>>>(Xq, Wq, bq, Qh, 0);
  proj_kernel<<<gblk, blk, 0, stream>>>(Xk, Wk, bk, Kh, 0);
  proj_kernel<<<gblk, blk, 0, stream>>>(Xv, Wv, bv, Vth, 1);
  flash_attn_kernel<<<dim3(SS / 64, BB * HH), dim3(128), 0, stream>>>(Qh, Kh, Vth, Ctx);
  outproj_kernel<<<gblk, blk, 0, stream>>>(Ctx, Wo, out);
}
